// SimpleRNNBase_1855425872458
// MI455X (gfx1250) — compile-verified, matrix operand order checked by reading, not run on hardware
//
#include <hip/hip_runtime.h>
#include <hip/hip_bf16.h>
#include <math.h>

// ---------------------------------------------------------------------------
// GRU-variant RNN for MI455X (gfx1250, wave32, WMMA).
//   T=2048, B=64, IN=H=256, GATE=3H (reset chunk provably unused -> skipped).
// Kernel 1: gi = input @ w_ih[256:768].T + b_ih[256:768]   (bf16 WMMA, f32 acc)
//           One block per M-tile; A-slice packed once into LDS (input read 1x
//           from HBM -> ~384MB total traffic, the bandwidth floor).
// Kernel 2: persistent single-workgroup scan; w_hh i/n slices pre-packed as
//           bf16 WMMA B-fragments in LDS (256KB), h state bf16 in LDS (32KB)
//           for the A-operand, f32 hidden state carried in REGISTERS.
// ---------------------------------------------------------------------------

typedef __attribute__((ext_vector_type(16))) __bf16 v16bf;
typedef __attribute__((ext_vector_type(8)))  float  v8f;

#define T_STEPS 2048
#define BB      64
#define HH      256
#define NGI     512          // i + n gate columns

union BFrag {
    uint4 q[2];
    v16bf v;
};

__device__ __forceinline__ v16bf pack_bf16(float4 a0, float4 a1, float4 a2, float4 a3) {
    v16bf v;
    v[0]  = (__bf16)a0.x; v[1]  = (__bf16)a0.y; v[2]  = (__bf16)a0.z; v[3]  = (__bf16)a0.w;
    v[4]  = (__bf16)a1.x; v[5]  = (__bf16)a1.y; v[6]  = (__bf16)a1.z; v[7]  = (__bf16)a1.w;
    v[8]  = (__bf16)a2.x; v[9]  = (__bf16)a2.y; v[10] = (__bf16)a2.z; v[11] = (__bf16)a2.w;
    v[12] = (__bf16)a3.x; v[13] = (__bf16)a3.y; v[14] = (__bf16)a3.z; v[15] = (__bf16)a3.w;
    return v;
}

// ---------------------------------------------------------------------------
// Kernel 1: one block (256 thr, 8 waves) per M-tile (16 rows of T*B=131072).
// Block packs A (16x256) once into LDS as WMMA fragments; each wave emits
// 4 of the 32 N-tiles. 8192 blocks.
// ---------------------------------------------------------------------------
__global__ __launch_bounds__(256) void gi_gemm_kernel(
    const float* __restrict__ in,     // [T*B][256]
    const float* __restrict__ w_ih,   // [768][256]
    const float* __restrict__ b_ih,   // [768]
    float* __restrict__ gi)           // [T*B][512]
{
    // A fragments: aB[ks][lane][e], 8*32*16 bf16 = 8KB
    __shared__ __bf16 aB[4096];

    const int tid  = threadIdx.x;
    const int lane = tid & 31;
    const int wave = tid >> 5;
    const int half = lane >> 4;
    const int nl   = lane & 15;
    const int tm   = blockIdx.x;                 // M tile

    // --- cooperative pack of the A slice (read input exactly once) ---------
    for (int idx = tid; idx < 4096; idx += 256) {
        const int e  = idx & 15;
        const int ln = (idx >> 4) & 31;
        const int ks = idx >> 9;
        const int kh = (ln >> 4) * 8;
        const int K  = ks * 32 + ((e < 8) ? (kh + e) : (16 + kh + (e - 8)));
        const int m  = tm * 16 + (ln & 15);
        aB[idx] = (__bf16)in[(size_t)m * 256 + K];
    }
    __syncthreads();

    // --- each wave: 4 N-tiles, A from LDS, B streamed from w_ih (L2) -------
    v8f c0 = {}, c1 = {}, c2 = {}, c3 = {};
#pragma unroll 1
    for (int ks = 0; ks < 8; ++ks) {
        BFrag fa;
        const __bf16* ap = aB + ((ks * 32 + lane) << 4);
        fa.q[0] = *(const uint4*)(ap);
        fa.q[1] = *(const uint4*)(ap + 8);

        const int kb = ks * 32 + half * 8;
#pragma unroll
        for (int q = 0; q < 4; ++q) {
            const int tn   = wave * 4 + q;
            const int wrow = 256 + tn * 16 + nl;
            const float* brow = w_ih + (size_t)wrow * 256;
            float4 b0 = *(const float4*)(brow + kb);
            float4 b1 = *(const float4*)(brow + kb + 4);
            float4 b2 = *(const float4*)(brow + kb + 16);
            float4 b3 = *(const float4*)(brow + kb + 20);
            const v16bf b = pack_bf16(b0, b1, b2, b3);
            if      (q == 0) c0 = __builtin_amdgcn_wmma_f32_16x16x32_bf16(false, fa.v, false, b, (short)0, c0, false, false);
            else if (q == 1) c1 = __builtin_amdgcn_wmma_f32_16x16x32_bf16(false, fa.v, false, b, (short)0, c1, false, false);
            else if (q == 2) c2 = __builtin_amdgcn_wmma_f32_16x16x32_bf16(false, fa.v, false, b, (short)0, c2, false, false);
            else             c3 = __builtin_amdgcn_wmma_f32_16x16x32_bf16(false, fa.v, false, b, (short)0, c3, false, false);
        }
    }

#pragma unroll
    for (int q = 0; q < 4; ++q) {
        const int   tn   = wave * 4 + q;
        const int   n    = tn * 16 + nl;
        const float bias = b_ih[256 + n];
        const v8f   c    = (q == 0) ? c0 : (q == 1) ? c1 : (q == 2) ? c2 : c3;
#pragma unroll
        for (int r = 0; r < 8; ++r) {
            const int mr = tm * 16 + r + half * 8;   // C layout: reg r -> M (+8 hi half)
            gi[(size_t)mr * NGI + n] = c[r] + bias;
        }
    }
}

// ---------------------------------------------------------------------------
// Kernel 2: persistent scan. 1 block x 1024 threads (32 waves) on one WGP.
// LDS: [0, 256KB)      w_hh i/n B-fragments (bf16, fragment-packed)
//      [256KB, 288KB)  h state bf16 [64][256] (A operand only)
// f32 hidden state lives in registers: the lane computing hy(b,j) is the lane
// that consumes h_prev(b,j) next step.
// ---------------------------------------------------------------------------
#define WB_BYTES   (8 * 32 * 32 * 16 * 2)   // 262144
#define LDS_BYTES  (WB_BYTES + BB * HH * 2) // 294912 = 288KB

extern __shared__ __attribute__((aligned(32))) char smem[];

__global__ __launch_bounds__(1024) void gru_scan_kernel(
    const float* __restrict__ gi,     // [T*B][512]
    const float* __restrict__ hx,     // [1][64][256]
    const float* __restrict__ w_hh,   // [768][256]
    const float* __restrict__ b_hh,   // [768]
    float* __restrict__ out)          // output [T][64][256] ++ h_last [64][256]
{
    __bf16* __restrict__ wB = (__bf16*)smem;               // packed weights
    __bf16* __restrict__ hB = (__bf16*)(smem + WB_BYTES);  // h state, bf16

    const int tid  = threadIdx.x;
    const int lane = tid & 31;
    const int wave = tid >> 5;
    const int half = lane >> 4;
    const int nl   = lane & 15;

    // --- setup: pack w_hh i/n slices into per-lane B fragments -------------
    for (int idx = tid; idx < 131072; idx += 1024) {
        const int e  = idx & 15;
        const int ln = (idx >> 4) & 31;
        const int gt = (idx >> 9) & 31;          // gate*16 + jt
        const int ks = idx >> 14;
        const int kh = (ln >> 4) * 8;
        const int K  = ks * 32 + ((e < 8) ? (kh + e) : (16 + kh + (e - 8)));
        const int j  = (gt & 15) * 16 + (ln & 15);
        const int wr = 256 + (gt >> 4) * 256 + j;           // skip reset rows
        wB[idx] = (__bf16)w_hh[(size_t)wr * 256 + K];
    }
    for (int idx = tid; idx < BB * HH; idx += 1024)
        hB[idx] = (__bf16)hx[idx];

    // --- tile ownership ----------------------------------------------------
    const int mt  = wave >> 3;          // M tile (0..3)
    const int jt0 = (wave * 2) & 15;    // first column tile
    const int am  = mt * 16 + nl;       // A row this lane feeds

    float bhi[2], bhn[2];
    float hv[2][8];                     // f32 hidden state, register-resident
#pragma unroll
    for (int p = 0; p < 2; ++p) {
        const int j = (jt0 + p) * 16 + nl;
        bhi[p] = b_hh[256 + j];
        bhn[p] = b_hh[512 + j];
#pragma unroll
        for (int r = 0; r < 8; ++r) {
            const int b = mt * 16 + r + half * 8;
            hv[p][r] = hx[b * 256 + j];
        }
    }
    __syncthreads();

#pragma unroll 1
    for (int t = 0; t < T_STEPS; ++t) {
        // prefetch next step's gi rows into L2 while this step computes
        if (t + 1 < T_STEPS) {
            const float* nxt = gi + ((size_t)(t + 1) * BB + mt * 16 + half * 8) * NGI
                                  + jt0 * 16 + nl;
            __builtin_prefetch(nxt, 0, 0);
        }

        v8f ci0 = {}, ci1 = {}, cn0 = {}, cn1 = {};

#pragma unroll
        for (int ks = 0; ks < 8; ++ks) {
            // A frag straight from bf16 LDS: 2x ds_load_b128, zero converts
            BFrag fa;
            const __bf16* ap = hB + am * 256 + ks * 32 + half * 8;
            fa.q[0] = *(const uint4*)(ap);
            fa.q[1] = *(const uint4*)(ap + 16);

#pragma unroll
            for (int p = 0; p < 2; ++p) {
                const int jt = jt0 + p;
                const __bf16* bi = wB + ((((ks * 32) + jt)      * 32 + lane) << 4);
                const __bf16* bn = wB + ((((ks * 32) + 16 + jt) * 32 + lane) << 4);
                BFrag fi, fn;
                fi.q[0] = *(const uint4*)(bi);
                fi.q[1] = *(const uint4*)(bi + 8);
                fn.q[0] = *(const uint4*)(bn);
                fn.q[1] = *(const uint4*)(bn + 8);
                if (p == 0) {
                    ci0 = __builtin_amdgcn_wmma_f32_16x16x32_bf16(
                            false, fa.v, false, fi.v, (short)0, ci0, false, false);
                    cn0 = __builtin_amdgcn_wmma_f32_16x16x32_bf16(
                            false, fa.v, false, fn.v, (short)0, cn0, false, false);
                } else {
                    ci1 = __builtin_amdgcn_wmma_f32_16x16x32_bf16(
                            false, fa.v, false, fi.v, (short)0, ci1, false, false);
                    cn1 = __builtin_amdgcn_wmma_f32_16x16x32_bf16(
                            false, fa.v, false, fn.v, (short)0, cn1, false, false);
                }
            }
        }

        // --- fused gate math (h_prev from registers) -----------------------
#pragma unroll
        for (int p = 0; p < 2; ++p) {
            const int j  = (jt0 + p) * 16 + nl;
            const v8f ci = (p == 0) ? ci0 : ci1;
            const v8f cn = (p == 0) ? cn0 : cn1;
#pragma unroll
            for (int r = 0; r < 8; ++r) {
                const int    b     = mt * 16 + r + half * 8;
                const size_t gbase = (size_t)t * BB + b;
                const float  gii   = gi[gbase * NGI + j];
                const float  gin   = gi[gbase * NGI + 256 + j];
                const float  ig    = 1.0f / (1.0f + __expf(-(gii + ci[r] + bhi[p])));
                const float  ng    = fmaxf(gin + cn[r] + bhn[p], 0.0f);
                const float  v     = ng + ig * (hv[p][r] - ng);
                out[gbase * HH + j] = v;
                hv[p][r] = v;
            }
        }
        __syncthreads();   // all A-fragment reads of h(t-1) complete
#pragma unroll
        for (int p = 0; p < 2; ++p) {
            const int j = (jt0 + p) * 16 + nl;
#pragma unroll
            for (int r = 0; r < 8; ++r) {
                const int b = mt * 16 + r + half * 8;
                hB[b * 256 + j] = (__bf16)hv[p][r];
            }
        }
        __syncthreads();   // h(t) visible to everyone
    }

    // --- h_last (exact f32 carry, from registers) --------------------------
    float* hlast = out + (size_t)T_STEPS * BB * HH;
#pragma unroll
    for (int p = 0; p < 2; ++p) {
        const int j = (jt0 + p) * 16 + nl;
#pragma unroll
        for (int r = 0; r < 8; ++r) {
            const int b = mt * 16 + r + half * 8;
            hlast[b * 256 + j] = hv[p][r];
        }
    }
}

// ---------------------------------------------------------------------------
extern "C" void kernel_launch(void* const* d_in, const int* in_sizes, int n_in,
                              void* d_out, int out_size, void* d_ws, size_t ws_size,
                              hipStream_t stream) {
    const float* in   = (const float*)d_in[0];   // [2048][64][256]
    const float* hx   = (const float*)d_in[1];   // [1][64][256]
    const float* w_ih = (const float*)d_in[2];   // [768][256]
    const float* w_hh = (const float*)d_in[3];   // [768][256]
    const float* b_ih = (const float*)d_in[4];   // [768]
    const float* b_hh = (const float*)d_in[5];   // [768]
    float* out = (float*)d_out;
    float* gi  = (float*)d_ws;                   // [2048*64][512] = 256 MB

    // Kernel 1: one block per M-tile (8192 x 256 threads)
    gi_gemm_kernel<<<8192, 256, 0, stream>>>(in, w_ih, b_ih, gi);

    // Kernel 2: persistent scan, 1 WGP, 288KB dynamic LDS
    gru_scan_kernel<<<1, 1024, LDS_BYTES, stream>>>(gi, hx, w_hh, b_hh, out);
}
